// GATModel_73804718015075
// MI455X (gfx1250) — matrix operand order repaired and checked by reading, not compiled, not run on hardware
//
#include <hip/hip_runtime.h>

typedef __attribute__((ext_vector_type(16))) _Float16 v16h;
typedef __attribute__((ext_vector_type(8)))  _Float16 v8h;
typedef __attribute__((ext_vector_type(8)))  float    v8f;

#define EMB  128
#define TDIM 32
#define EDIM 64

// ---- ordered-uint encoding for float atomicMax (sign-magnitude -> total order) ----
__device__ __forceinline__ unsigned f2ord(float f) {
  unsigned u = __float_as_uint(f);
  return (u & 0x80000000u) ? ~u : (u | 0x80000000u);
}
__device__ __forceinline__ float ord2f(unsigned u) {
  return (u & 0x80000000u) ? __uint_as_float(u & 0x7fffffffu) : __uint_as_float(~u);
}

// ---------------- embedding gather: x[n,:] = emb_table[node_ids[n],:] ----------------
__global__ void k_gather_emb(const int* __restrict__ ids,
                             const float* __restrict__ table,
                             float* __restrict__ x, int nrows) {
  int node = blockIdx.x * (blockDim.x >> 5) + (threadIdx.x >> 5);
  int lane = threadIdx.x & 31;
  if (node >= nrows) return;
  const float4* src = (const float4*)(table + (size_t)ids[node] * EMB);
  ((float4*)(x + (size_t)node * EMB))[lane] = src[lane];
}

// ---------------- v64[i,k] = sum_j eW[i,k,j] * a_edge[i,j]  (4*64 dots) ----------------
__global__ void k_eproj(const float* __restrict__ eW, const float* __restrict__ aeg,
                        float* __restrict__ v64) {
  int idx = threadIdx.x;          // 256 threads, one (i,k) each
  int i = idx >> 6, k = idx & 63;
  float s = 0.f;
  const float* row = eW + ((size_t)i * EDIM + k) * EMB;
  const float* av  = aeg + (size_t)i * EMB;
  for (int j = 0; j < EMB; ++j) s += row[j] * av[j];
  v64[idx] = s;
}

// ------- per-edge fused time-encoding + alpha_edge dots for all 4 convs -------
__global__ void k_edge_feat(const float* __restrict__ t, const float* __restrict__ eattr,
                            const float* __restrict__ tw, const float* __restrict__ tb,
                            const float* __restrict__ v64, float* __restrict__ ae, int E) {
  __shared__ float s_tw[TDIM], s_tb[TDIM], s_v[256];
  if (threadIdx.x < TDIM) { s_tw[threadIdx.x] = tw[threadIdx.x]; s_tb[threadIdx.x] = tb[threadIdx.x]; }
  s_v[threadIdx.x] = v64[threadIdx.x];
  __syncthreads();
  int e = blockIdx.x * blockDim.x + threadIdx.x;
  if (e >= E) return;
  float tv = t[e];
  float acc0 = 0.f, acc1 = 0.f, acc2 = 0.f, acc3 = 0.f;
  #pragma unroll 4
  for (int k = 0; k < TDIM; ++k) {
    float te = __cosf(tv * s_tw[k] + s_tb[k]);
    acc0 += te * s_v[0 * 64 + k];  acc1 += te * s_v[1 * 64 + k];
    acc2 += te * s_v[2 * 64 + k];  acc3 += te * s_v[3 * 64 + k];
  }
  const float4* ep = (const float4*)(eattr + (size_t)e * TDIM);
  #pragma unroll
  for (int q = 0; q < 8; ++q) {
    float4 v = ep[q];
    int k = q * 4;
    acc0 += v.x * s_v[0*64+32+k]   + v.y * s_v[0*64+32+k+1] + v.z * s_v[0*64+32+k+2] + v.w * s_v[0*64+32+k+3];
    acc1 += v.x * s_v[1*64+32+k]   + v.y * s_v[1*64+32+k+1] + v.z * s_v[1*64+32+k+2] + v.w * s_v[1*64+32+k+3];
    acc2 += v.x * s_v[2*64+32+k]   + v.y * s_v[2*64+32+k+1] + v.z * s_v[2*64+32+k+2] + v.w * s_v[2*64+32+k+3];
    acc3 += v.x * s_v[3*64+32+k]   + v.y * s_v[3*64+32+k+1] + v.z * s_v[3*64+32+k+2] + v.w * s_v[3*64+32+k+3];
  }
  ae[0 * (size_t)E + e] = acc0;  ae[1 * (size_t)E + e] = acc1;
  ae[2 * (size_t)E + e] = acc2;  ae[3 * (size_t)E + e] = acc3;
}

// ---- W (f32, [conv][k][n]) -> f16 transposed WhT[conv][n][k], once per launch ----
__global__ void k_cvt_wT(const float* __restrict__ W, _Float16* __restrict__ WhT, int total) {
  int i = blockIdx.x * blockDim.x + threadIdx.x;
  if (i >= total) return;
  int conv = i >> 14;            // / (128*128)
  int rem  = i & 16383;
  int n = rem >> 7, k = rem & 127;
  WhT[i] = (_Float16)W[((size_t)conv << 14) + (size_t)k * EMB + n];
}

// ---------------- H = X @ W via v_wmma_f32_16x16x32_f16 ----------------
// block = 256 threads = 8 waves; block owns a 16-row slab, wave w owns cols [16w,16w+16)
__global__ void k_gemm_wmma(const float* __restrict__ X, const _Float16* __restrict__ WhT,
                            float* __restrict__ H, int nrows) {
  __shared__ __align__(16) _Float16 As[16 * EMB];   // 4 KB  : A tile, row-major f16
  __shared__ __align__(16) _Float16 Bs[EMB * EMB];  // 32 KB : Bs[n][k] = W[k][n] f16
  int tid = threadIdx.x;
  int m0 = blockIdx.x * 16;

  // stage A: 2048 halves, 8 contiguous per thread (2x float4 load -> 1x v8h store)
  {
    int idx8 = tid * 8;
    int r = idx8 >> 7, c = idx8 & 127;
    int row = m0 + r;
    v8h av = {};
    if (row < nrows) {
      const float4* xp = (const float4*)(X + (size_t)row * EMB + c);
      float4 x0 = xp[0], x1 = xp[1];
      av[0] = (_Float16)x0.x; av[1] = (_Float16)x0.y; av[2] = (_Float16)x0.z; av[3] = (_Float16)x0.w;
      av[4] = (_Float16)x1.x; av[5] = (_Float16)x1.y; av[6] = (_Float16)x1.z; av[7] = (_Float16)x1.w;
    }
    *(v8h*)&As[idx8] = av;
  }
  // stage B: 16384 halves = 2048 float4 chunks, straight copy
  {
    const float4* src = (const float4*)WhT;
    float4* dst = (float4*)Bs;
    #pragma unroll
    for (int i = tid; i < 2048; i += 256) dst[i] = src[i];
  }
  __syncthreads();

  int lane = tid & 31;
  int n0 = (tid >> 5) * 16;           // wave's column tile
  int mr   = lane & 15;               // A row / B column within tile
  int half = lane >> 4;               // lane-half selector
  v8f c = {};
  #pragma unroll
  for (int kk = 0; kk < 4; ++kk) {    // K = 128 in chunks of 32
    int kbase = kk * 32;
    // A 16x32 f16, contiguous k order (same as B): lane-half h holds K = {h*16..h*16+15}
    v8h alo = *(const v8h*)&As[mr * EMB + kbase + half * 16];
    v8h ahi = *(const v8h*)&As[mr * EMB + kbase + half * 16 + 8];
    v16h a = __builtin_shufflevector(alo, ahi, 0,1,2,3,4,5,6,7,8,9,10,11,12,13,14,15);
    // B 32x16 f16, contiguous k order (same as A): lane-half h holds K = {h*16..h*16+15}, col = lane&15
    // Bs is [n][k] so the 16-half fragment is contiguous.
    v16h b = *(const v16h*)&Bs[(size_t)(n0 + mr) * EMB + kbase + half * 16];
    c = __builtin_amdgcn_wmma_f32_16x16x32_f16(false, a, false, b, (short)0, c, false, false);
  }
  // C/D: VGPR r -> row r (lanes<16) / r+8 (lanes>=16). Hoist the bounds check:
  // full tiles (the only case when nrows % 16 == 0) take one unguarded store loop.
  float* __restrict__ hp = H + ((size_t)(m0 + half * 8) * EMB + n0 + mr);
  if (m0 + 16 <= nrows) {
    #pragma unroll
    for (int r = 0; r < 8; ++r) hp[(size_t)r * EMB] = c[r];
  } else {
    int rmax = nrows - m0 - half * 8;
    #pragma unroll
    for (int r = 0; r < 8; ++r) if (r < rmax) hp[(size_t)r * EMB] = c[r];
  }
}

// ---------------- per-node dots with a_src / a_dst (wave per node) ----------------
__global__ void k_node_dots(const float* __restrict__ H, const float* __restrict__ as,
                            const float* __restrict__ ad, float* __restrict__ ds,
                            float* __restrict__ dd, int nrows) {
  int node = blockIdx.x * (blockDim.x >> 5) + (threadIdx.x >> 5);
  int lane = threadIdx.x & 31;
  if (node >= nrows) return;
  float4 hv = ((const float4*)(H + (size_t)node * EMB))[lane];
  float4 av = ((const float4*)as)[lane];
  float4 bv = ((const float4*)ad)[lane];
  float s = hv.x * av.x + hv.y * av.y + hv.z * av.z + hv.w * av.w;
  float d = hv.x * bv.x + hv.y * bv.y + hv.z * bv.z + hv.w * bv.w;
  #pragma unroll
  for (int off = 16; off > 0; off >>= 1) {
    s += __shfl_down(s, off, 32);
    d += __shfl_down(d, off, 32);
  }
  if (lane == 0) { ds[node] = s; dd[node] = d; }
}

// ---------------- per-stage init: acc=0, denom=0, amax=-1e30 ----------------
__global__ void k_init(float* acc0, float* acc1, unsigned* amax0, unsigned* amax1,
                       float* den0, float* den1, int nrows) {
  int i = blockIdx.x * blockDim.x + threadIdx.x;
  size_t tot = (size_t)nrows * EMB;
  if ((size_t)i < tot) { acc0[i] = 0.f; acc1[i] = 0.f; }
  if (i < nrows) {
    den0[i] = 0.f; den1[i] = 0.f;
    unsigned m = f2ord(-1e30f);
    amax0[i] = m; amax1[i] = m;
  }
}

// ---------------- edge pass 1: alpha + leaky relu + segment max ----------------
__global__ void k_edge_alpha(const int* __restrict__ ei, const int* __restrict__ et,
                             const float* __restrict__ ds0, const float* __restrict__ dd0,
                             const float* __restrict__ ds1, const float* __restrict__ dd1,
                             const float* __restrict__ ae0, const float* __restrict__ ae1,
                             float* __restrict__ alpha, unsigned* __restrict__ amax0,
                             unsigned* __restrict__ amax1, int E) {
  int e = blockIdx.x * blockDim.x + threadIdx.x;
  if (e >= E) return;
  int s = ei[e], d = ei[E + e];
  int ty = et[e];
  float a = (ty == 0) ? (ds0[s] + dd0[d] + ae0[e]) : (ds1[s] + dd1[d] + ae1[e]);
  a = (a > 0.f) ? a : 0.2f * a;          // leaky_relu(0.2)
  alpha[e] = a;
  atomicMax((ty == 0) ? &amax0[d] : &amax1[d], f2ord(a));
}

// ---------------- edge pass 2: w = exp(alpha - amax), segment sum ----------------
__global__ void k_edge_w(const int* __restrict__ ei, const int* __restrict__ et,
                         float* __restrict__ alpha, const unsigned* __restrict__ amax0,
                         const unsigned* __restrict__ amax1, float* __restrict__ den0,
                         float* __restrict__ den1, int E) {
  int e = blockIdx.x * blockDim.x + threadIdx.x;
  if (e >= E) return;
  int d = ei[E + e];
  int ty = et[e];
  float am = ord2f((ty == 0) ? amax0[d] : amax1[d]);
  float w = __expf(alpha[e] - am);
  alpha[e] = w;
  atomicAdd((ty == 0) ? &den0[d] : &den1[d], w);
}

// ---- edge pass 3: acc[dst] += w/denom * h[src]  (wave per edge, grid-stride, prefetch) ----
__global__ void k_edge_scatter(const int* __restrict__ ei, const int* __restrict__ et,
                               const float* __restrict__ w,
                               const float* __restrict__ den0, const float* __restrict__ den1,
                               const float* __restrict__ h0, const float* __restrict__ h1,
                               float* __restrict__ acc0, float* __restrict__ acc1, int E) {
  int lane = threadIdx.x & 31;
  int wid  = blockIdx.x * (blockDim.x >> 5) + (threadIdx.x >> 5);
  int nw   = gridDim.x * (blockDim.x >> 5);
  for (int e = wid; e < E; e += nw) {
    int en = e + nw;                       // prefetch next edge's source row (512B)
    if (en < E) {
      int sn = ei[en];
      const float* hn = (et[en] == 0) ? h0 : h1;
      __builtin_prefetch(hn + (size_t)sn * EMB + lane * 4, 0, 0);
    }
    int s = ei[e], d = ei[E + e];
    int ty = et[e];
    float den = (ty == 0) ? den0[d] : den1[d];
    float coeff = w[e] / (den + 1e-16f);
    const float* h = (ty == 0) ? h0 : h1;
    float* acc = ((ty == 0) ? acc0 : acc1) + (size_t)d * EMB + lane * 4;
    float4 hv = ((const float4*)(h + (size_t)s * EMB))[lane];
    atomicAdd(acc + 0, hv.x * coeff);
    atomicAdd(acc + 1, hv.y * coeff);
    atomicAdd(acc + 2, hv.z * coeff);
    atomicAdd(acc + 3, hv.w * coeff);
  }
}

// ---------------- epilogue: bias + relu + node-type select ----------------
__global__ void k_combine(const float* __restrict__ acc0, const float* __restrict__ acc1,
                          const float* __restrict__ b0, const float* __restrict__ b1,
                          const int* __restrict__ ntype, float* __restrict__ out, int nrows) {
  int i = blockIdx.x * blockDim.x + threadIdx.x;
  if ((size_t)i >= (size_t)nrows * EMB) return;
  int n = i >> 7, j = i & 127;
  float r0 = fmaxf(acc0[i] + b0[j], 0.f);   // resource-conv result (mask edge_type==0)
  float r1 = fmaxf(acc1[i] + b1[j], 0.f);   // user-conv result (mask edge_type==1)
  out[i] = (ntype[n] == 0) ? r1 : r0;       // users take the r2u conv
}

extern "C" void kernel_launch(void* const* d_in, const int* in_sizes, int n_in,
                              void* d_out, int out_size, void* d_ws, size_t ws_size,
                              hipStream_t stream) {
  const int*   node_ids  = (const int*)d_in[0];
  const int*   node_type = (const int*)d_in[1];
  const int*   edge_index= (const int*)d_in[2];
  const int*   edge_type = (const int*)d_in[3];
  const float* edge_attr = (const float*)d_in[4];
  const float* t         = (const float*)d_in[5];
  const float* emb       = (const float*)d_in[6];
  const float* tw        = (const float*)d_in[7];
  const float* tb        = (const float*)d_in[8];
  const float* convW     = (const float*)d_in[9];
  const float* asrc      = (const float*)d_in[10];
  const float* adst      = (const float*)d_in[11];
  const float* eW        = (const float*)d_in[12];
  const float* aeg       = (const float*)d_in[13];
  const float* bias      = (const float*)d_in[14];
  const int N = in_sizes[0];
  const int E = in_sizes[3];

  // ---- workspace layout (16B-aligned buffers first) ----
  float* ws = (float*)d_ws;
  size_t off = 0;
  float* xcur  = ws + off; off += (size_t)N * EMB;
  float* h0    = ws + off; off += (size_t)N * EMB;
  float* h1    = ws + off; off += (size_t)N * EMB;
  float* acc0  = ws + off; off += (size_t)N * EMB;
  float* acc1  = ws + off; off += (size_t)N * EMB;
  float* aedge = ws + off; off += 4 * (size_t)E;
  float* alpha = ws + off; off += (size_t)E;
  float* ds0   = ws + off; off += N;
  float* dd0   = ws + off; off += N;
  float* ds1   = ws + off; off += N;
  float* dd1   = ws + off; off += N;
  float* den0  = ws + off; off += N;
  float* den1  = ws + off; off += N;
  unsigned* amax0 = (unsigned*)(ws + off); off += N;
  unsigned* amax1 = (unsigned*)(ws + off); off += N;
  float* v64   = ws + off; off += 256;
  _Float16* WhT = (_Float16*)(ws + off);  // 4*128*128 halves (transposed, f16)

  const int waves_pb = 8;                 // 256-thread blocks = 8 waves
  dim3 blk(256);
  int gN128 = (int)(((size_t)N * EMB + 255) / 256);
  int gE    = (E + 255) / 256;
  int gNw   = (N + waves_pb - 1) / waves_pb;
  int gEw   = (E + 4 * waves_pb - 1) / (4 * waves_pb);  // 4 edges per wave (grid-stride)
  int gM    = (N + 15) / 16;

  // ---- prologue ----
  k_gather_emb<<<gNw, blk, 0, stream>>>(node_ids, emb, xcur, N);
  k_eproj<<<1, blk, 0, stream>>>(eW, aeg, v64);
  k_edge_feat<<<gE, blk, 0, stream>>>(t, edge_attr, tw, tb, v64, aedge, E);
  k_cvt_wT<<<(4 * EMB * EMB + 255) / 256, blk, 0, stream>>>(convW, WhT, 4 * EMB * EMB);

  // ---- two stages, each = {conv(2s, mask0), conv(2s+1, mask1)} ----
  for (int stage = 0; stage < 2; ++stage) {
    int i0 = 2 * stage, i1 = i0 + 1;
    const float* xin = xcur;
    float* xout = (stage == 0) ? xcur : (float*)d_out;

    k_gemm_wmma<<<gM, blk, 0, stream>>>(xin, WhT + (size_t)i0 * EMB * EMB, h0, N);
    k_gemm_wmma<<<gM, blk, 0, stream>>>(xin, WhT + (size_t)i1 * EMB * EMB, h1, N);

    k_node_dots<<<gNw, blk, 0, stream>>>(h0, asrc + (size_t)i0 * EMB, adst + (size_t)i0 * EMB, ds0, dd0, N);
    k_node_dots<<<gNw, blk, 0, stream>>>(h1, asrc + (size_t)i1 * EMB, adst + (size_t)i1 * EMB, ds1, dd1, N);

    k_init<<<gN128, blk, 0, stream>>>(acc0, acc1, amax0, amax1, den0, den1, N);

    k_edge_alpha<<<gE, blk, 0, stream>>>(edge_index, edge_type, ds0, dd0, ds1, dd1,
                                         aedge + (size_t)i0 * E, aedge + (size_t)i1 * E,
                                         alpha, amax0, amax1, E);
    k_edge_w<<<gE, blk, 0, stream>>>(edge_index, edge_type, alpha, amax0, amax1, den0, den1, E);
    k_edge_scatter<<<gEw, blk, 0, stream>>>(edge_index, edge_type, alpha, den0, den1,
                                            h0, h1, acc0, acc1, E);

    k_combine<<<gN128, blk, 0, stream>>>(acc0, acc1, bias + (size_t)i0 * EMB,
                                         bias + (size_t)i1 * EMB, node_type, xout, N);
  }
}
